// SimpleGRU_52080773431935
// MI455X (gfx1250) — matrix operand order repaired and checked by reading, not compiled, not run on hardware
//
#include <hip/hip_runtime.h>

#define BATCH 128
#define SEQ 512
#define ISZ 512
#define HSZ 1024

typedef __attribute__((ext_vector_type(16))) __bf16 v16bf;
typedef __attribute__((ext_vector_type(8)))  float  v8f;

union AFrag { v16bf v; uint4 q[2]; };

__device__ __forceinline__ unsigned short f2bf(float f) {
  unsigned int u = __builtin_bit_cast(unsigned int, f);
  u += 0x7FFFu + ((u >> 16) & 1u);           // round-to-nearest-even
  return (unsigned short)(u >> 16);
}

__device__ __forceinline__ v8f wmma_bf16(v16bf a, v16bf b, v8f c) {
  return __builtin_amdgcn_wmma_f32_16x16x32_bf16(false, a, false, b, (short)0, c,
                                                 false, false);
}

// B fragment from fragment-major packed weights: tile = 512 bf16, lane -> 32B chunk.
__device__ __forceinline__ v16bf load_b(const unsigned short* __restrict__ tile,
                                        int lane) {
  return *(const v16bf*)(tile + lane * 16);
}

__device__ __forceinline__ v16bf load_a_frag(const unsigned short* __restrict__ p) {
  AFrag f;
  f.q[0] = *(const uint4*)(p);
  f.q[1] = *(const uint4*)(p + 8);
  return f.v;
}

// ---------- prep kernels ----------
__global__ void cast_bf16_kernel(const float* __restrict__ src,
                                 unsigned short* __restrict__ dst, size_t n) {
  size_t i = (size_t)blockIdx.x * blockDim.x + threadIdx.x;
  if (i < n) dst[i] = f2bf(src[i]);
}

// Pack row-major W[K][N] (fp32) into WMMA-B fragment-major bf16:
// element e (0..15) of lane L in tile (kc,nt) holds W[kc*32 + (L<16?0:16) + e][nt*16 + (L&15)].
__global__ void pack_w_kernel(const float* __restrict__ W,
                              unsigned short* __restrict__ P, int K, int N) {
  size_t p = (size_t)blockIdx.x * blockDim.x + threadIdx.x;
  if (p >= (size_t)K * N) return;
  int e       = (int)(p & 15);
  int lane    = (int)((p >> 4) & 31);
  size_t tile = p >> 9;
  int ntN = N >> 4;
  int kc  = (int)(tile / ntN);
  int nt  = (int)(tile % ntN);
  int k = kc * 32 + ((lane >> 4) * 16) + e;
  int n = nt * 16 + (lane & 15);
  P[p] = f2bf(W[(size_t)k * N + n]);
}

__global__ void init_h_kernel(const float* __restrict__ h0,
                              float* __restrict__ hf,
                              unsigned short* __restrict__ hbf, int n) {
  int i = blockIdx.x * blockDim.x + threadIdx.x;
  if (i < n) { float v = h0[i]; hf[i] = v; hbf[i] = f2bf(v); }
}

// ---------- recurrent step, stage A: R and Z gates ----------
// 2 n-tiles x 2 gates per wave -> 4 independent WMMA chains; depth-2 SW pipeline.
#define NT_A 2
__global__ __launch_bounds__(128) void gru_gates_rz(
    const unsigned short* __restrict__ hbf,   // [B,H] bf16
    const float*          __restrict__ hf,    // [B,H] f32
    const unsigned short* __restrict__ xbf,   // [B,T,I] bf16
    const unsigned short* __restrict__ Whr,   // packed [H/32][H/16][512]
    const unsigned short* __restrict__ Whz,
    const unsigned short* __restrict__ Wxr,   // packed [I/32][H/16][512]
    const unsigned short* __restrict__ Wxz,
    const float* __restrict__ br, const float* __restrict__ bz,
    unsigned short* __restrict__ Rh,          // [B,H] bf16 out
    float*          __restrict__ Zbuf,        // [B,H] f32 out
    int t) {
  const int lane = threadIdx.x & 31;
  const int gw   = blockIdx.x * (blockDim.x >> 5) + (threadIdx.x >> 5); // 0..255
  const int mt  = gw >> 5;                 // 0..7
  const int nt0 = (gw & 31) * NT_A;        // 0..62 step 2
  const int mBase = mt * 16;
  const int ntN = HSZ >> 4;
  const int c0  = (lane >> 4) * 16;        // A-frag K sub-offset per lane group

  v8f accR[NT_A] = {};
  v8f accZ[NT_A] = {};

  v16bf aP, aQ;
  v16bf bP0[NT_A], bP1[NT_A], bQ0[NT_A], bQ1[NT_A];

  auto loadC = [&](const unsigned short* aRow, const unsigned short* w0,
                   const unsigned short* w1, int c,
                   v16bf& a, v16bf (&b0)[NT_A], v16bf (&b1)[NT_A]) {
    a = load_a_frag(aRow + c * 32);
    const size_t tb = ((size_t)(c * ntN + nt0)) << 9;
#pragma unroll
    for (int j = 0; j < NT_A; ++j) {
      b0[j] = load_b(w0 + tb + ((size_t)j << 9), lane);
      b1[j] = load_b(w1 + tb + ((size_t)j << 9), lane);
    }
  };
  auto mac = [&](v16bf a, v16bf (&b0)[NT_A], v16bf (&b1)[NT_A]) {
#pragma unroll
    for (int j = 0; j < NT_A; ++j) {
      accR[j] = wmma_bf16(a, b0[j], accR[j]);
      accZ[j] = wmma_bf16(a, b1[j], accZ[j]);
    }
  };

  // ---- phase 1: h @ W_h{r,z}, K = H (32 chunks) ----
  {
    const unsigned short* aRow = hbf + (size_t)(mBase + (lane & 15)) * HSZ + c0;
    const int KH = HSZ / 32;
    loadC(aRow, Whr, Whz, 0, aP, bP0, bP1);
    loadC(aRow, Whr, Whz, 1, aQ, bQ0, bQ1);
    for (int c = 0; c < KH - 2; c += 2) {
      mac(aP, bP0, bP1);
      loadC(aRow, Whr, Whz, c + 2, aP, bP0, bP1);
      mac(aQ, bQ0, bQ1);
      loadC(aRow, Whr, Whz, c + 3, aQ, bQ0, bQ1);
    }
    mac(aP, bP0, bP1);
    mac(aQ, bQ0, bQ1);
  }
  // ---- phase 2: x_t @ W_x{r,z}, K = I (16 chunks, fused input projection) ----
  {
    const unsigned short* xRow =
        xbf + ((size_t)(mBase + (lane & 15)) * SEQ + t) * ISZ + c0;
    const int KX = ISZ / 32;
    loadC(xRow, Wxr, Wxz, 0, aP, bP0, bP1);
    loadC(xRow, Wxr, Wxz, 1, aQ, bQ0, bQ1);
    for (int c = 0; c < KX - 2; c += 2) {
      mac(aP, bP0, bP1);
      loadC(xRow, Wxr, Wxz, c + 2, aP, bP0, bP1);
      mac(aQ, bQ0, bQ1);
      loadC(xRow, Wxr, Wxz, c + 3, aQ, bQ0, bQ1);
    }
    mac(aP, bP0, bP1);
    mac(aQ, bQ0, bQ1);
  }

  // epilogue: C/D layout -> lane gives n = lane&15, VGPR r gives m = r + (lane<16?0:8)
  const int mo = mBase + ((lane >> 4) * 8);
#pragma unroll
  for (int j = 0; j < NT_A; ++j) {
    const int n = (nt0 + j) * 16 + (lane & 15);
    const float brv = br[n], bzv = bz[n];
#pragma unroll
    for (int r = 0; r < 8; ++r) {
      const int idx = (mo + r) * HSZ + n;
      float rg = 1.0f / (1.0f + __expf(-(accR[j][r] + brv)));
      float zg = 1.0f / (1.0f + __expf(-(accZ[j][r] + bzv)));
      Rh[idx]   = f2bf(rg * hf[idx]);
      Zbuf[idx] = zg;
    }
  }
}

// ---------- recurrent step, stage B: candidate + state update ----------
// 4 n-tiles per wave -> 4 independent WMMA chains; depth-2 SW pipeline.
#define NT_B 4
__global__ __launch_bounds__(128) void gru_update(
    const unsigned short* __restrict__ Rh,    // [B,H] bf16
    const unsigned short* __restrict__ xbf,   // [B,T,I] bf16
    const unsigned short* __restrict__ Whh,
    const unsigned short* __restrict__ Wxh,
    const float* __restrict__ bh,
    const float* __restrict__ Zbuf,
    float* __restrict__ hf,                   // [B,H] f32 (read old / write new)
    unsigned short* __restrict__ hbf,         // [B,H] bf16 (write new)
    float* __restrict__ out,                  // [B,T,H]
    int t) {
  const int lane = threadIdx.x & 31;
  const int gw   = blockIdx.x * (blockDim.x >> 5) + (threadIdx.x >> 5); // 0..127
  const int mt  = gw >> 4;                 // 0..7
  const int nt0 = (gw & 15) * NT_B;        // 0..60 step 4
  const int mBase = mt * 16;
  const int ntN = HSZ >> 4;
  const int c0  = (lane >> 4) * 16;

  v8f acc[NT_B] = {};

  v16bf aP, aQ;
  v16bf bP[NT_B], bQ[NT_B];

  auto loadC = [&](const unsigned short* aRow, const unsigned short* w, int c,
                   v16bf& a, v16bf (&b)[NT_B]) {
    a = load_a_frag(aRow + c * 32);
    const size_t tb = ((size_t)(c * ntN + nt0)) << 9;
#pragma unroll
    for (int j = 0; j < NT_B; ++j) b[j] = load_b(w + tb + ((size_t)j << 9), lane);
  };
  auto mac = [&](v16bf a, v16bf (&b)[NT_B]) {
#pragma unroll
    for (int j = 0; j < NT_B; ++j) acc[j] = wmma_bf16(a, b[j], acc[j]);
  };

  // ---- phase 1: (R*h) @ W_hh, K = H ----
  {
    const unsigned short* aRow = Rh + (size_t)(mBase + (lane & 15)) * HSZ + c0;
    const int KH = HSZ / 32;
    loadC(aRow, Whh, 0, aP, bP);
    loadC(aRow, Whh, 1, aQ, bQ);
    for (int c = 0; c < KH - 2; c += 2) {
      mac(aP, bP);
      loadC(aRow, Whh, c + 2, aP, bP);
      mac(aQ, bQ);
      loadC(aRow, Whh, c + 3, aQ, bQ);
    }
    mac(aP, bP);
    mac(aQ, bQ);
  }
  // ---- phase 2: x_t @ W_xh, K = I ----
  {
    const unsigned short* xRow =
        xbf + ((size_t)(mBase + (lane & 15)) * SEQ + t) * ISZ + c0;
    const int KX = ISZ / 32;
    loadC(xRow, Wxh, 0, aP, bP);
    loadC(xRow, Wxh, 1, aQ, bQ);
    for (int c = 0; c < KX - 2; c += 2) {
      mac(aP, bP);
      loadC(xRow, Wxh, c + 2, aP, bP);
      mac(aQ, bQ);
      loadC(xRow, Wxh, c + 3, aQ, bQ);
    }
    mac(aP, bP);
    mac(aQ, bQ);
  }

  const int mo = mBase + ((lane >> 4) * 8);
#pragma unroll
  for (int j = 0; j < NT_B; ++j) {
    const int n = (nt0 + j) * 16 + (lane & 15);
    const float bhv = bh[n];
#pragma unroll
    for (int r = 0; r < 8; ++r) {
      const int m   = mo + r;
      const int idx = m * HSZ + n;
      float htil = tanhf(acc[j][r] + bhv);
      float z    = Zbuf[idx];
      float hnew = z * htil + (1.0f - z) * hf[idx];
      out[((size_t)m * SEQ + t) * HSZ + n] = hnew;
      hf[idx]  = hnew;
      hbf[idx] = f2bf(hnew);
    }
  }
}

extern "C" void kernel_launch(void* const* d_in, const int* in_sizes, int n_in,
                              void* d_out, int out_size, void* d_ws, size_t ws_size,
                              hipStream_t stream) {
  const float* x   = (const float*)d_in[0];
  const float* h0  = (const float*)d_in[1];
  const float* Whr = (const float*)d_in[2];
  const float* Wxr = (const float*)d_in[3];
  const float* br  = (const float*)d_in[4];
  const float* Whz = (const float*)d_in[5];
  const float* Wxz = (const float*)d_in[6];
  const float* bz  = (const float*)d_in[7];
  const float* Whh = (const float*)d_in[8];
  const float* Wxh = (const float*)d_in[9];
  const float* bh  = (const float*)d_in[10];
  float* out = (float*)d_out;

  char* ws = (char*)d_ws;
  size_t off = 0;
  auto alloc = [&](size_t bytes) -> void* {
    void* p = ws + off;
    off = (off + bytes + 255) & ~(size_t)255;
    return p;
  };

  unsigned short* xbf  = (unsigned short*)alloc((size_t)BATCH * SEQ * ISZ * 2);
  unsigned short* WxrP = (unsigned short*)alloc((size_t)ISZ * HSZ * 2);
  unsigned short* WxzP = (unsigned short*)alloc((size_t)ISZ * HSZ * 2);
  unsigned short* WxhP = (unsigned short*)alloc((size_t)ISZ * HSZ * 2);
  unsigned short* WhrP = (unsigned short*)alloc((size_t)HSZ * HSZ * 2);
  unsigned short* WhzP = (unsigned short*)alloc((size_t)HSZ * HSZ * 2);
  unsigned short* WhhP = (unsigned short*)alloc((size_t)HSZ * HSZ * 2);
  float*          hf   = (float*)alloc((size_t)BATCH * HSZ * 4);
  unsigned short* hbf  = (unsigned short*)alloc((size_t)BATCH * HSZ * 2);
  unsigned short* Rh   = (unsigned short*)alloc((size_t)BATCH * HSZ * 2);
  float*          Zb   = (float*)alloc((size_t)BATCH * HSZ * 4);

  // one-time conversions / packing
  {
    size_t n = (size_t)BATCH * SEQ * ISZ;
    cast_bf16_kernel<<<(unsigned)((n + 255) / 256), 256, 0, stream>>>(x, xbf, n);
  }
  pack_w_kernel<<<(ISZ * HSZ) / 256, 256, 0, stream>>>(Wxr, WxrP, ISZ, HSZ);
  pack_w_kernel<<<(ISZ * HSZ) / 256, 256, 0, stream>>>(Wxz, WxzP, ISZ, HSZ);
  pack_w_kernel<<<(ISZ * HSZ) / 256, 256, 0, stream>>>(Wxh, WxhP, ISZ, HSZ);
  pack_w_kernel<<<(HSZ * HSZ) / 256, 256, 0, stream>>>(Whr, WhrP, HSZ, HSZ);
  pack_w_kernel<<<(HSZ * HSZ) / 256, 256, 0, stream>>>(Whz, WhzP, HSZ, HSZ);
  pack_w_kernel<<<(HSZ * HSZ) / 256, 256, 0, stream>>>(Whh, WhhP, HSZ, HSZ);
  init_h_kernel<<<(BATCH * HSZ) / 256, 256, 0, stream>>>(h0, hf, hbf, BATCH * HSZ);

  // sequential scan (graph-captured): stage A = 256 waves, stage B = 128 waves
  for (int t = 0; t < SEQ; ++t) {
    gru_gates_rz<<<64, 128, 0, stream>>>(hbf, hf, xbf, WhrP, WhzP, WxrP, WxzP,
                                         br, bz, Rh, Zb, t);
    gru_update<<<32, 128, 0, stream>>>(Rh, xbf, WhhP, WxhP, bh, Zb,
                                       hf, hbf, out, t);
  }
}
